// GraphNet_85031762526342
// MI455X (gfx1250) — matrix operand order repaired and checked by reading, not compiled, not run on hardware
//
#include <hip/hip_runtime.h>
#include <hip/hip_bf16.h>
#include <math.h>

// ---------------- problem constants (match reference) ----------------
#define NNODES 400000
#define NEDGES 800000
#define NGRAPH 16384
#define MAXDEG 10   // deg_idx in [0, 10]

typedef __attribute__((ext_vector_type(16))) __bf16 v16bf;
typedef __attribute__((ext_vector_type(8)))  float  v8f;

// ---------------- graph kernels ----------------

__global__ void k_degree(const int* __restrict__ dst, unsigned* __restrict__ deg, int E) {
    int e = blockIdx.x * blockDim.x + threadIdx.x;
    if (e < E) atomicAdd(&deg[dst[e]], 1u);
}

__global__ void k_clamp_deg(int* __restrict__ deg, int n) {
    int i = blockIdx.x * blockDim.x + threadIdx.x;
    if (i < n) deg[i] = min(deg[i], MAXDEG);
}

// hsum[dst[e], f] += x[src[e], f]    (F = 1<<logF features)
__global__ void k_scatter_add(const float* __restrict__ x, const int* __restrict__ src,
                              const int* __restrict__ dst, float* __restrict__ hsum,
                              int E, int logF) {
    long long t = (long long)blockIdx.x * blockDim.x + threadIdx.x;
    int F = 1 << logF;
    if (t >= ((long long)E << logF)) return;
    int e = (int)(t >> logF);
    int f = (int)(t & (F - 1));
    atomicAdd(&hsum[((size_t)dst[e] << logF) + f], x[((size_t)src[e] << logF) + f]);
}

// out[n,o] = (Wl[d] @ hsum[n])[o] + bl[d,o] + (Wr[d] @ x[n])[o], optional relu.
// Weights are [11, 32, Fin] row-major; 45KB total -> L2 resident.
__global__ void k_mfconv_apply(const float* __restrict__ hsum, const float* __restrict__ xs,
                               const int* __restrict__ deg_idx,
                               const float* __restrict__ Wl, const float* __restrict__ bl,
                               const float* __restrict__ Wr, float* __restrict__ out,
                               int n_nodes, int Fin, int do_relu) {
    int t = blockIdx.x * blockDim.x + threadIdx.x;
    if (t >= n_nodes * 32) return;
    int n = t >> 5, o = t & 31;
    int d = deg_idx[n];
    const float* wl = Wl + ((size_t)d * 32 + o) * Fin;
    const float* wr = Wr + ((size_t)d * 32 + o) * Fin;
    const float* h  = hsum + (size_t)n * Fin;
    const float* xv = xs   + (size_t)n * Fin;
    float acc = bl[d * 32 + o];
    for (int i = 0; i < Fin; ++i)
        acc = fmaf(wl[i], h[i], fmaf(wr[i], xv[i], acc));
    out[(size_t)n * 32 + o] = do_relu ? fmaxf(acc, 0.0f) : acc;
}

// pooled[batch[n], f] += h[n, f]; cnt[batch[n]] += 1
__global__ void k_pool(const float* __restrict__ h, const int* __restrict__ batch,
                       float* __restrict__ pooled, float* __restrict__ cnt, int n_nodes) {
    int t = blockIdx.x * blockDim.x + threadIdx.x;
    if (t >= n_nodes * 32) return;
    int n = t >> 5, f = t & 31;
    int b = batch[n];
    atomicAdd(&pooled[(size_t)b * 32 + f], h[(size_t)n * 32 + f]);
    if (f == 0) atomicAdd(&cnt[b], 1.0f);
}

// ---------------- WMMA operand packing (CDNA5 VGPR layouts, CDNA5 ISA §7.12.2) ----------
// A (16x32 bf16 tile): lane&15 = M row; lanes 0-15 hold K {0..7,16..23},
//                      lanes 16-31 hold K {8..15,24..31}; element pairs per VGPR.
// Packed so that [ (rb*nkb + kb)*32 + lane ] is one contiguous v16bf (32 bytes).
__global__ void k_pack_A(const float* __restrict__ A, __bf16* __restrict__ out,
                         int M, int K, int nkb) {
    long long t = (long long)blockIdx.x * blockDim.x + threadIdx.x;
    long long total = (long long)(M >> 4) * nkb * 512;
    if (t >= total) return;
    int e    = (int)(t & 15);
    int lane = (int)((t >> 4) & 31);
    long long blk = t >> 9;            // rb*nkb + kb
    int kb = (int)(blk % nkb);
    int rb = (int)(blk / nkb);
    int m = (rb << 4) + (lane & 15);
    int k = (kb << 5) + ((lane >> 4) << 3) + e + ((e >= 8) ? 8 : 0);
    float v = (k < K) ? A[(size_t)m * K + k] : 0.0f;
    out[t] = (__bf16)v;
}

// B (32x16 bf16 tile) from W[N,K] row-major (B = W^T tile): lane&15 = N col, K order same as A.
// Packed so that [ (nb*nkb + kb)*32 + lane ] is one contiguous v16bf.
__global__ void k_pack_B(const float* __restrict__ W, __bf16* __restrict__ out,
                         int N, int K, int nkb) {
    long long t = (long long)blockIdx.x * blockDim.x + threadIdx.x;
    long long total = (long long)(N >> 4) * nkb * 512;
    if (t >= total) return;
    int e    = (int)(t & 15);
    int lane = (int)((t >> 4) & 31);
    long long blk = t >> 9;            // nb*nkb + kb
    int kb = (int)(blk % nkb);
    int nb = (int)(blk / nkb);
    int n = (nb << 4) + (lane & 15);
    int k = (kb << 5) + ((lane >> 4) << 3) + e + ((e >= 8) ? 8 : 0);
    float v = (k < K) ? W[(size_t)n * K + k] : 0.0f;
    out[t] = (__bf16)v;
}

// ---------------- bf16 WMMA GEMM on packed operands ----------------
// One wave computes a 16x64 C strip: A operand reused across 4 B tiles / 4 accumulators.
// Inner loop per K-block: 1x v16bf A load + 4x v16bf B loads + 4x v_wmma. Branch-free.
__global__ void k_gemm_wmma(const v16bf* __restrict__ Apk, const v16bf* __restrict__ Bpk,
                            const float* __restrict__ bias, float* __restrict__ C,
                            int M, int N, int nkb, int do_relu) {
    int wave = (int)((blockIdx.x * (size_t)blockDim.x + threadIdx.x) >> 5);
    int lane = threadIdx.x & 31;
    int tilesN = N >> 6;               // 64-column super-tiles
    int total  = (M >> 4) * tilesN;
    if (wave >= total) return;
    int rb  = wave / tilesN;           // 16-row block
    int nb0 = (wave % tilesN) << 2;    // first of four 16-col tiles

    v8f acc0 = {}, acc1 = {}, acc2 = {}, acc3 = {};
    const v16bf* ap = Apk + (size_t)rb * nkb * 32 + lane;
    const v16bf* bp = Bpk + lane;

    for (int kb = 0; kb < nkb; ++kb) {
        __builtin_prefetch(ap + (size_t)(kb + 1) * 32, 0, 1);   // global_prefetch_b8
        v16bf a  = ap[(size_t)kb * 32];
        v16bf b0 = bp[((size_t)(nb0 + 0) * nkb + kb) * 32];
        v16bf b1 = bp[((size_t)(nb0 + 1) * nkb + kb) * 32];
        v16bf b2 = bp[((size_t)(nb0 + 2) * nkb + kb) * 32];
        v16bf b3 = bp[((size_t)(nb0 + 3) * nkb + kb) * 32];
        acc0 = __builtin_amdgcn_wmma_f32_16x16x32_bf16(false, a, false, b0, (short)0, acc0, false, false);
        acc1 = __builtin_amdgcn_wmma_f32_16x16x32_bf16(false, a, false, b1, (short)0, acc1, false, false);
        acc2 = __builtin_amdgcn_wmma_f32_16x16x32_bf16(false, a, false, b2, (short)0, acc2, false, false);
        acc3 = __builtin_amdgcn_wmma_f32_16x16x32_bf16(false, a, false, b3, (short)0, acc3, false, false);
    }

    // C/D f32 layout: VGPR r -> row = rb*16 + r + 8*(lane>=16); col = tile + (lane&15)
    int row_off = (rb << 4) + ((lane >> 4) << 3);
    int cbase   = (nb0 << 4) + (lane & 15);
    v8f accs[4] = {acc0, acc1, acc2, acc3};
#pragma unroll
    for (int j = 0; j < 4; ++j) {
        int col = cbase + (j << 4);
        float bi = bias[col];
#pragma unroll
        for (int r = 0; r < 8; ++r) {
            float v = accs[j][r] + bi;
            if (do_relu) v = fmaxf(v, 0.0f);
            C[(size_t)(row_off + r) * N + col] = v;
        }
    }
}

// ---------------- BatchNorm (training stats, biased var) ----------------

__global__ void k_bn_stats(const float* __restrict__ X, int M, int N,
                           float* __restrict__ mu, float* __restrict__ rs) {
    int c = blockIdx.x;                 // one column per block
    float s = 0.0f, s2 = 0.0f;
    for (int r = threadIdx.x; r < M; r += blockDim.x) {
        float v = X[(size_t)r * N + c];
        s += v; s2 += v * v;
    }
    __shared__ float sh[256], sh2[256];
    sh[threadIdx.x] = s; sh2[threadIdx.x] = s2;
    __syncthreads();
    for (int off = blockDim.x >> 1; off > 0; off >>= 1) {
        if (threadIdx.x < off) {
            sh[threadIdx.x]  += sh[threadIdx.x + off];
            sh2[threadIdx.x] += sh2[threadIdx.x + off];
        }
        __syncthreads();
    }
    if (threadIdx.x == 0) {
        float m   = sh[0] / (float)M;
        float var = sh2[0] / (float)M - m * m;
        mu[c] = m;
        rs[c] = rsqrtf(var + 1e-5f);
    }
}

__global__ void k_bn_apply(float* __restrict__ X, int M, int N,
                           const float* __restrict__ mu, const float* __restrict__ rs,
                           const float* __restrict__ g, const float* __restrict__ be) {
    long long t = (long long)blockIdx.x * blockDim.x + threadIdx.x;
    if (t >= (long long)M * N) return;
    int c = (int)(t % N);
    float v = X[t];
    X[t] = (v - mu[c]) * rs[c] * g[c] + be[c];
}

// ---------------- head: concat([pooled/cnt, m3]) @ W_out^T + b, sigmoid ----------------

__global__ void k_head(const float* __restrict__ pooled, const float* __restrict__ cnt,
                       const float* __restrict__ m3, const float* __restrict__ Wout,
                       const float* __restrict__ bout, float* __restrict__ out, int B) {
    int b = blockIdx.x * blockDim.x + threadIdx.x;
    if (b >= B) return;
    float inv = 1.0f / fmaxf(cnt[b], 1.0f);
    float acc = bout[0];
    for (int j = 0; j < 32; ++j) acc = fmaf(pooled[(size_t)b * 32 + j] * inv, Wout[j], acc);
    for (int j = 0; j < 64; ++j) acc = fmaf(m3[(size_t)b * 64 + j], Wout[32 + j], acc);
    out[b] = 1.0f / (1.0f + expf(-acc));
}

// ---------------- launch ----------------

static inline int cdiv(long long a, int b) { return (int)((a + b - 1) / b); }

extern "C" void kernel_launch(void* const* d_in, const int* in_sizes, int n_in,
                              void* d_out, int out_size, void* d_ws, size_t ws_size,
                              hipStream_t stream) {
    // inputs (setup_inputs dict order)
    const float* x     = (const float*)d_in[0];                 // [N,8]
    const int*   src   = (const int*)d_in[1];                   // edge_index[0]
    const int*   dst   = ((const int*)d_in[1]) + NEDGES;        // edge_index[1]
    const int*   batch = (const int*)d_in[2];                   // [N]
    const float* xmord = (const float*)d_in[3];                 // [B,862]
    const float* Wl1 = (const float*)d_in[4];  const float* bl1 = (const float*)d_in[5];
    const float* Wr1 = (const float*)d_in[6];
    const float* Wl2 = (const float*)d_in[7];  const float* bl2 = (const float*)d_in[8];
    const float* Wr2 = (const float*)d_in[9];
    const float* Wl3 = (const float*)d_in[10]; const float* bl3 = (const float*)d_in[11];
    const float* Wr3 = (const float*)d_in[12];
    const float* Wfc1 = (const float*)d_in[13]; const float* bfc1 = (const float*)d_in[14];
    const float* g1   = (const float*)d_in[15]; const float* be1  = (const float*)d_in[16];
    const float* Wfc2 = (const float*)d_in[17]; const float* bfc2 = (const float*)d_in[18];
    const float* g2   = (const float*)d_in[19]; const float* be2  = (const float*)d_in[20];
    const float* Wfc3 = (const float*)d_in[21]; const float* bfc3 = (const float*)d_in[22];
    const float* g3   = (const float*)d_in[23]; const float* be3  = (const float*)d_in[24];
    const float* Wout = (const float*)d_in[25]; const float* bout = (const float*)d_in[26];
    float* out = (float*)d_out;

    // workspace carve (~657 MB)
    size_t off = 0;
    char* base = (char*)d_ws;
    auto carve = [&](size_t bytes) -> void* {
        void* p = base + off;
        off += (bytes + 255) & ~(size_t)255;
        return p;
    };
    int*    degIdx = (int*)   carve((size_t)NNODES * 4);
    float*  bufA   = (float*) carve((size_t)NNODES * 32 * 4);   // neighbor sums
    float*  bufB   = (float*) carve((size_t)NNODES * 32 * 4);   // h1 / h3
    float*  bufC   = (float*) carve((size_t)NNODES * 32 * 4);   // h2
    float*  pooled = (float*) carve((size_t)NGRAPH * 32 * 4);
    float*  cnt    = (float*) carve((size_t)NGRAPH * 4);
    float*  m1     = (float*) carve((size_t)NGRAPH * 512 * 4);
    float*  m2     = (float*) carve((size_t)NGRAPH * 128 * 4);
    float*  m3     = (float*) carve((size_t)NGRAPH * 64 * 4);
    float*  mu     = (float*) carve(512 * 4);
    float*  rs     = (float*) carve(512 * 4);
    __bf16* Apk    = (__bf16*)carve((size_t)NGRAPH * 27 * 32 * 2 * 16); // M * max nkb * 512 bf16
    __bf16* Bpk    = (__bf16*)carve((size_t)512 * 27 * 32 * 2);        // maxN/16 * max nkb * 512 bf16
    (void)ws_size; (void)n_in; (void)in_sizes; (void)out_size;

    const int T = 256;

    // ---- degrees ----
    hipMemsetAsync(degIdx, 0, (size_t)NNODES * 4, stream);
    k_degree<<<cdiv(NEDGES, T), T, 0, stream>>>(dst, (unsigned*)degIdx, NEDGES);
    k_clamp_deg<<<cdiv(NNODES, T), T, 0, stream>>>(degIdx, NNODES);

    // ---- MFConv layer 1 (Fin=8) ----
    hipMemsetAsync(bufA, 0, (size_t)NNODES * 8 * 4, stream);
    k_scatter_add<<<cdiv((long long)NEDGES * 8, T), T, 0, stream>>>(x, src, dst, bufA, NEDGES, 3);
    k_mfconv_apply<<<cdiv((long long)NNODES * 32, T), T, 0, stream>>>(
        bufA, x, degIdx, Wl1, bl1, Wr1, bufB, NNODES, 8, 1);

    // ---- MFConv layer 2 (Fin=32) ----
    hipMemsetAsync(bufA, 0, (size_t)NNODES * 32 * 4, stream);
    k_scatter_add<<<cdiv((long long)NEDGES * 32, T), T, 0, stream>>>(bufB, src, dst, bufA, NEDGES, 5);
    k_mfconv_apply<<<cdiv((long long)NNODES * 32, T), T, 0, stream>>>(
        bufA, bufB, degIdx, Wl2, bl2, Wr2, bufC, NNODES, 32, 1);

    // ---- MFConv layer 3 (Fin=32, no relu) ----
    hipMemsetAsync(bufA, 0, (size_t)NNODES * 32 * 4, stream);
    k_scatter_add<<<cdiv((long long)NEDGES * 32, T), T, 0, stream>>>(bufC, src, dst, bufA, NEDGES, 5);
    k_mfconv_apply<<<cdiv((long long)NNODES * 32, T), T, 0, stream>>>(
        bufA, bufC, degIdx, Wl3, bl3, Wr3, bufB, NNODES, 32, 0);

    // ---- global mean pool ----
    hipMemsetAsync(pooled, 0, (size_t)NGRAPH * 32 * 4, stream);
    hipMemsetAsync(cnt,    0, (size_t)NGRAPH * 4,      stream);
    k_pool<<<cdiv((long long)NNODES * 32, T), T, 0, stream>>>(bufB, batch, pooled, cnt, NNODES);

    // ---- MLP fc layers via packed-operand WMMA GEMM ----
    struct Layer { const float* A; const float* W; const float* b;
                   const float* g; const float* be; float* O; int N; int K; };
    Layer layers[3] = {
        { xmord, Wfc1, bfc1, g1, be1, m1, 512, 862 },
        { m1,    Wfc2, bfc2, g2, be2, m2, 128, 512 },
        { m2,    Wfc3, bfc3, g3, be3, m3,  64, 128 },
    };
    for (int L = 0; L < 3; ++L) {
        int N = layers[L].N, K = layers[L].K;
        int nkb = (K + 31) >> 5;
        long long aTot = (long long)(NGRAPH >> 4) * nkb * 512;
        long long bTot = (long long)(N >> 4) * nkb * 512;
        k_pack_A<<<cdiv(aTot, T), T, 0, stream>>>(layers[L].A, Apk, NGRAPH, K, nkb);
        k_pack_B<<<cdiv(bTot, T), T, 0, stream>>>(layers[L].W, Bpk, N, K, nkb);
        long long waves = (long long)(NGRAPH >> 4) * (N >> 6);
        k_gemm_wmma<<<cdiv(waves * 32, T), T, 0, stream>>>(
            (const v16bf*)Apk, (const v16bf*)Bpk, layers[L].b, layers[L].O,
            NGRAPH, N, nkb, 1);
        k_bn_stats<<<N, 256, 0, stream>>>(layers[L].O, NGRAPH, N, mu, rs);
        k_bn_apply<<<cdiv((long long)NGRAPH * N, T), T, 0, stream>>>(
            layers[L].O, NGRAPH, N, mu, rs, layers[L].g, layers[L].be);
    }

    // ---- concat + linear + sigmoid ----
    k_head<<<cdiv(NGRAPH, T), T, 0, stream>>>(pooled, cnt, m3, Wout, bout, out, NGRAPH);
}
